// CrossImageAttention_69767448756720
// MI455X (gfx1250) — matrix operand order repaired and checked by reading, not compiled, not run on hardware
//
#include <hip/hip_runtime.h>
#include <stdint.h>

typedef __attribute__((ext_vector_type(16))) __bf16 v16bf;
typedef __attribute__((ext_vector_type(8)))  __bf16 v8bf;
typedef __attribute__((ext_vector_type(8)))  float  v8f;

#define C_DIM   768
#define N_POS   4096
#define B_DIM   8
#define H_HEADS 8
#define D_HEAD  96
#define NC      (N_POS * C_DIM)        // 3145728
#define BNC     (B_DIM * NC)           // 25165824
#define CC      (C_DIM * C_DIM)        // 589824

// ---------------- fp32 <-> bf16 helpers (RNE) ----------------
__device__ __forceinline__ unsigned short f2bf(float f) {
  union { float f; uint32_t u; } c; c.f = f;
  uint32_t u = c.u;
  uint32_t r = u + 0x7FFFu + ((u >> 16) & 1u);
  if ((u & 0x7F800000u) == 0x7F800000u) r = u;   // Inf/NaN passthrough
  return (unsigned short)(r >> 16);
}
__device__ __forceinline__ float bf2f(uint32_t hbits) {
  union { uint32_t u; float f; } c; c.u = hbits << 16;
  return c.f;
}
__device__ __forceinline__ uint32_t pack2(float a, float b) {
  return (uint32_t)f2bf(a) | ((uint32_t)f2bf(b) << 16);
}

// ---------------- CDNA5 async global->LDS copy (ASYNCcnt-tracked) ----------------
// IOFFSET is added to BOTH the LDS and the global address, so one base register
// covers both 16B halves of a thread's 32B slice.
__device__ __forceinline__ void async_ld32B(uint32_t lds_addr, const unsigned short* g) {
  asm volatile("global_load_async_to_lds_b128 %0, %1, off\n\t"
               "global_load_async_to_lds_b128 %0, %1, off offset:16"
               :: "v"(lds_addr), "v"(g) : "memory");
}
__device__ __forceinline__ void wait_async0() {
  asm volatile("s_wait_asynccnt 0x0" ::: "memory");
}
__device__ __forceinline__ uint32_t lds_addr_of(const void* p) {
  return (uint32_t)(uintptr_t)p;   // LDS aperture keeps wave-relative offset in addr[31:0]
}

// ---------------- prep kernels ----------------
__global__ __launch_bounds__(256) void k_cvt4(const float4* __restrict__ s,
                                              uint2* __restrict__ d, int n4) {
  int i = blockIdx.x * blockDim.x + threadIdx.x;
  if (i >= n4) return;
  float4 v = s[i];
  uint2 o; o.x = pack2(v.x, v.y); o.y = pack2(v.z, v.w);
  d[i] = o;
}

// convert support to bf16 AND produce bf16 of sum_b support[b] (for ksum GEMM)
__global__ __launch_bounds__(256) void k_support_prep(const float2* __restrict__ sup,
                                                      uint32_t* __restrict__ sbf,
                                                      uint32_t* __restrict__ ssum) {
  int i = blockIdx.x * blockDim.x + threadIdx.x;   // over NC/2
  if (i >= NC / 2) return;
  float ax = 0.f, ay = 0.f;
#pragma unroll
  for (int b = 0; b < B_DIM; ++b) {
    float2 v = sup[(size_t)b * (NC / 2) + i];
    ax += v.x; ay += v.y;
    sbf[(size_t)b * (NC / 2) + i] = pack2(v.x, v.y);
  }
  ssum[i] = pack2(ax, ay);
}

// ---------------- bf16 WMMA GEMM:  C[M,768] = A[M,768] @ W[768,768]^T + bias ----------------
#define BM 128
#define BN 128
#define BK 32
#define LDK 40                      // padded LDS k-stride (bf16 elems) -> conflict-free frag reads
#define KTILES (C_DIM / BK)         // 24

__device__ __forceinline__ v16bf mk16(v8bf lo, v8bf hi) {
  v16bf r;
#pragma unroll
  for (int i = 0; i < 8; ++i) { r[i] = lo[i]; r[i + 8] = hi[i]; }
  return r;
}
__device__ __forceinline__ void store_out(float* p, float v) { *p = v; }
__device__ __forceinline__ void store_out(unsigned short* p, float v) { *p = f2bf(v); }

template <typename OutT>
__global__ __launch_bounds__(256) void k_gemm_bf16(
    const unsigned short* __restrict__ A,   // M x 768 bf16, row-major
    const unsigned short* __restrict__ W,   // 768 x 768 bf16, row-major (out_ch, in_ch)
    const float* __restrict__ bias,         // 768
    OutT* __restrict__ Cout,                // M x 768
    int M, float bias_scale) {
  __shared__ __align__(16) unsigned short As[2][BM * LDK];
  __shared__ __align__(16) unsigned short Bs[2][BN * LDK];

  const int tid  = threadIdx.x;
  const int lane = tid & 31;
  const int wave = tid >> 5;
  const int wm   = wave & 3;          // wave M offset = wm*32
  const int wn   = wave >> 2;         // wave N offset = wn*64
  const int mbase = blockIdx.x * BM;
  const int nbase = blockIdx.y * BN;

  // global staging: 2 threads per tile row, 16 bf16 (32B) per thread, async -> LDS
  const int grow = tid >> 1;
  const int gk   = (tid & 1) * 16;
  const unsigned short* Ag = A + (size_t)(mbase + grow) * C_DIM + gk;
  const unsigned short* Wg = W + (size_t)(nbase + grow) * C_DIM + gk;
  const int sOff = grow * LDK + gk;
  const uint32_t asAddr[2] = { lds_addr_of(&As[0][sOff]), lds_addr_of(&As[1][sOff]) };
  const uint32_t bsAddr[2] = { lds_addr_of(&Bs[0][sOff]), lds_addr_of(&Bs[1][sOff]) };

  const v8f vzero = {0.f, 0.f, 0.f, 0.f, 0.f, 0.f, 0.f, 0.f};
  v8f acc[2][4];
#pragma unroll
  for (int mi = 0; mi < 2; ++mi)
#pragma unroll
    for (int ni = 0; ni < 4; ++ni) acc[mi][ni] = vzero;

  // per-lane fragment read offsets (ISA 7.12.2 layouts)
  const int fr = lane & 15;
  const int fh = lane >> 4;
  const int aOff0 = (wm * 32 + fr) * LDK + fh * 8;    // A: chunks at +0 and +16 (K split)
  const int bOff0 = (wn * 64 + fr) * LDK + fh * 8;    // B: chunks at +0 and +16 (K split)

  // stage tile 0 into buffer 0
  async_ld32B(asAddr[0], Ag);
  async_ld32B(bsAddr[0], Wg);
  wait_async0();
  __syncthreads();

  for (int kt = 0; kt < KTILES; ++kt) {
    const int buf = kt & 1;
    if (kt + 1 < KTILES) {                       // async-stage next K-tile into other buffer
      async_ld32B(asAddr[buf ^ 1], Ag + (kt + 1) * BK);
      async_ld32B(bsAddr[buf ^ 1], Wg + (kt + 1) * BK);
    }

    v16bf afrag[2], bfrag[4];
#pragma unroll
    for (int mi = 0; mi < 2; ++mi) {
      const unsigned short* p = &As[buf][aOff0 + mi * 16 * LDK];
      afrag[mi] = mk16(*(const v8bf*)p, *(const v8bf*)(p + 16));
    }
#pragma unroll
    for (int ni = 0; ni < 4; ++ni) {
      const unsigned short* p = &Bs[buf][bOff0 + ni * 16 * LDK];
      bfrag[ni] = mk16(*(const v8bf*)p, *(const v8bf*)(p + 16));
    }
#pragma unroll
    for (int mi = 0; mi < 2; ++mi)
#pragma unroll
      for (int ni = 0; ni < 4; ++ni)
        acc[mi][ni] = __builtin_amdgcn_wmma_f32_16x16x32_bf16(
            false, afrag[mi], false, bfrag[ni], (short)0, acc[mi][ni], false, false);

    if (kt + 1 < KTILES) {
      wait_async0();          // staged tile fully in LDS before making it visible
      __syncthreads();
    }
  }

  // epilogue: C/D frag layout -> row = r + 8*fh, col = fr
#pragma unroll
  for (int mi = 0; mi < 2; ++mi) {
#pragma unroll
    for (int ni = 0; ni < 4; ++ni) {
      const int col = nbase + wn * 64 + ni * 16 + fr;
      const float bv = bias[col] * bias_scale;
#pragma unroll
      for (int r = 0; r < 8; ++r) {
        const int row = mbase + wm * 32 + mi * 16 + fh * 8 + r;
        store_out(&Cout[(size_t)row * C_DIM + col], acc[mi][ni][r] + bv);
      }
    }
  }
}

// ---------------- fused attention: logits -> softmax -> @V -> permuted bf16 store ----------------
__global__ __launch_bounds__(256) void k_attn(const unsigned short* __restrict__ q,   // (B,N,C) bf16
                                              const float* __restrict__ ksum,         // (N,C) f32
                                              const unsigned short* __restrict__ v,   // (B,N,C) bf16
                                              unsigned short* __restrict__ perm) {    // (B,N,C) bf16
  const int t = blockIdx.x * blockDim.x + threadIdx.x;   // B*N*h threads
  const int i = t & 7;
  const int n = (t >> 3) & (N_POS - 1);
  const int b = t >> 15;

  const unsigned short* qrow = q + ((size_t)b * N_POS + n) * C_DIM + i * D_HEAD;
  const float* kb = ksum + (size_t)n * C_DIM;

  float logit[H_HEADS];
#pragma unroll
  for (int j = 0; j < H_HEADS; ++j) logit[j] = 0.f;
  for (int dd = 0; dd < D_HEAD; dd += 2) {
    const uint32_t qp = *(const uint32_t*)(qrow + dd);
    const float q0 = bf2f(qp & 0xFFFFu), q1 = bf2f(qp >> 16);
#pragma unroll
    for (int j = 0; j < H_HEADS; ++j)
      logit[j] += q0 * kb[j * D_HEAD + dd] + q1 * kb[j * D_HEAD + dd + 1];
  }
  const float s = 9.7979589711327124f;   // sqrt(96): reference divides by d^-0.5
  float mx = logit[0] * s;
#pragma unroll
  for (int j = 1; j < H_HEADS; ++j) mx = fmaxf(mx, logit[j] * s);
  float p[H_HEADS]; float sum = 0.f;
#pragma unroll
  for (int j = 0; j < H_HEADS; ++j) { p[j] = __expf(logit[j] * s - mx); sum += p[j]; }
  const float inv = 1.f / sum;
#pragma unroll
  for (int j = 0; j < H_HEADS; ++j) p[j] *= inv;

  const unsigned short* vb = v + ((size_t)b * N_POS + n) * C_DIM;
  // torch view/permute scramble: m = n*8+i -> hh = m>>12, n2 = m&4095, channel hh*96+dd
  const int m2 = (n << 3) | i;
  const int hh = m2 >> 12;
  const int n2 = m2 & (N_POS - 1);
  unsigned short* dst = perm + ((size_t)b * N_POS + n2) * C_DIM + hh * D_HEAD;
  for (int dd = 0; dd < D_HEAD; dd += 2) {
    float o0 = 0.f, o1 = 0.f;
#pragma unroll
    for (int j = 0; j < H_HEADS; ++j) {
      const uint32_t pv = *(const uint32_t*)(vb + j * D_HEAD + dd);
      o0 += p[j] * bf2f(pv & 0xFFFFu);
      o1 += p[j] * bf2f(pv >> 16);
    }
    *(uint32_t*)(dst + dd) = pack2(o0, o1);
  }
}

// ---------------- host orchestration ----------------
extern "C" void kernel_launch(void* const* d_in, const int* in_sizes, int n_in,
                              void* d_out, int out_size, void* d_ws, size_t ws_size,
                              hipStream_t stream) {
  (void)n_in; (void)out_size; (void)ws_size;
  if (in_sizes[0] != BNC || in_sizes[2] != CC) return;  // shape guard (constant per harness)

  const float* query   = (const float*)d_in[0];
  const float* support = (const float*)d_in[1];
  const float* Wq = (const float*)d_in[2]; const float* bq = (const float*)d_in[3];
  const float* Wk = (const float*)d_in[4]; const float* bk = (const float*)d_in[5];
  const float* Wv = (const float*)d_in[6]; const float* bv = (const float*)d_in[7];
  const float* Wp = (const float*)d_in[8]; const float* bp = (const float*)d_in[9];
  float* out = (float*)d_out;

  unsigned short* ws = (unsigned short*)d_ws;
  unsigned short* qbf    = ws;                     // BNC bf16
  unsigned short* sbf    = qbf    + (size_t)BNC;   // BNC bf16
  unsigned short* ssumbf = sbf    + (size_t)BNC;   // NC  bf16
  unsigned short* wqb    = ssumbf + (size_t)NC;    // CC  bf16
  unsigned short* wkb    = wqb    + (size_t)CC;
  unsigned short* wvb    = wkb    + (size_t)CC;
  unsigned short* wpb    = wvb    + (size_t)CC;
  unsigned short* qout   = wpb    + (size_t)CC;    // BNC bf16
  unsigned short* vout   = qout   + (size_t)BNC;   // BNC bf16
  unsigned short* permb  = vout   + (size_t)BNC;   // BNC bf16
  float* ksumf = (float*)(permb + (size_t)BNC);    // NC  f32

  const int M = B_DIM * N_POS;   // 32768

  k_cvt4<<<(BNC / 4) / 256, 256, 0, stream>>>((const float4*)query, (uint2*)qbf, BNC / 4);
  k_cvt4<<<(CC / 4) / 256, 256, 0, stream>>>((const float4*)Wq, (uint2*)wqb, CC / 4);
  k_cvt4<<<(CC / 4) / 256, 256, 0, stream>>>((const float4*)Wk, (uint2*)wkb, CC / 4);
  k_cvt4<<<(CC / 4) / 256, 256, 0, stream>>>((const float4*)Wv, (uint2*)wvb, CC / 4);
  k_cvt4<<<(CC / 4) / 256, 256, 0, stream>>>((const float4*)Wp, (uint2*)wpb, CC / 4);
  k_support_prep<<<(NC / 2) / 256, 256, 0, stream>>>((const float2*)support,
                                                     (uint32_t*)sbf, (uint32_t*)ssumbf);

  dim3 g1(M / BM, C_DIM / BN);        // 256 x 6
  dim3 g2(N_POS / BM, C_DIM / BN);    // 32  x 6
  // q = query @ Wq^T + bq
  k_gemm_bf16<unsigned short><<<g1, 256, 0, stream>>>(qbf, wqb, bq, qout, M, 1.0f);
  // v = support @ Wv^T + bv
  k_gemm_bf16<unsigned short><<<g1, 256, 0, stream>>>(sbf, wvb, bv, vout, M, 1.0f);
  // ksum = (sum_b support_b) @ Wk^T + B*bk   (linearity of the projection)
  k_gemm_bf16<float><<<g2, 256, 0, stream>>>(ssumbf, wkb, bk, ksumf, N_POS, (float)B_DIM);

  k_attn<<<(M * H_HEADS) / 256, 256, 0, stream>>>(qout, ksumf, vout, permb);

  // out = perm @ Wp^T + bp   (fp32 into d_out)
  k_gemm_bf16<float><<<g1, 256, 0, stream>>>(permb, wpb, bp, out, M, 1.0f);
}
